// Decoder_53541062312229
// MI455X (gfx1250) — matrix operand order repaired and checked by reading, not compiled, not run on hardware
//
#include <hip/hip_runtime.h>
#include <hip/hip_bf16.h>

// ---------------- problem constants ----------------
constexpr int T = 50, B = 64, S = 400, V = 50000;
constexpr int D = 512, E = 512, H = 512, AA = 512;   // A renamed AA
constexpr int G3 = 3 * H;                            // 1536
constexpr int KX = D + E;                            // 1024 (gru input)
constexpr int KQ = H + E;                            // 1024 (query input)
constexpr int KR = D + H + E + E;                    // 2048 (readout input)
constexpr int HP = H / 2;                            // 256 (maxout)

// output layout (floats), concatenated in reference return order
constexpr size_t OUT_GOUT   = 0;                                   // [T,B,HP]
constexpr size_t OUT_HID    = OUT_GOUT + (size_t)T * B * HP;       // [B,H]
constexpr size_t OUT_ALAST  = OUT_HID + (size_t)B * H;             // [B,S]
constexpr size_t OUT_CCTX   = OUT_ALAST + (size_t)B * S;           // [B,E]
constexpr size_t OUT_PG     = OUT_CCTX + (size_t)B * E;            // [T,B]
constexpr size_t OUT_ATTN   = OUT_PG + (size_t)T * B;              // [T,B,S]
constexpr size_t OUT_CLOSS  = OUT_ATTN + (size_t)T * B * S;        // [T,B]
constexpr size_t OUT_COVF   = OUT_CLOSS + (size_t)T * B;           // [B,S]

typedef __attribute__((ext_vector_type(8)))  float  v8f;
typedef __attribute__((ext_vector_type(8)))  __bf16 v8bf;
typedef __attribute__((ext_vector_type(16))) __bf16 v16bf;

__device__ __forceinline__ __bf16 f2bf(float f) {
  union { float f; unsigned u; } in; in.f = f;
  unsigned u = in.u;
  unsigned r = u + 0x7fffu + ((u >> 16) & 1u);      // RNE
  union { unsigned short s; __bf16 b; } out;
  out.s = (unsigned short)(r >> 16);
  return out.b;
}
__device__ __forceinline__ float sigf(float x) { return 1.0f / (1.0f + __expf(-x)); }

// ---------------- WMMA bf16 GEMM: C[M,N] = A[M,K] * BT[N,K]^T + bias ----------------
// One wave computes a 16x64 C block (4 x 16x16 tiles): the A fragment is loaded once
// per K-step and reused across 4 B fragments -> 2.5 b128 loads per v_wmma instead of 4,
// and 4x less global A traffic. Requires M%16==0, N%64==0, K%32==0 (true for all calls).
__global__ void gemm_bf16_wmma4(const __bf16* __restrict__ A, const __bf16* __restrict__ BT,
                                const float* __restrict__ bias, float* __restrict__ C,
                                int M, int N, int K) {
  int wave = (int)((blockIdx.x * blockDim.x + threadIdx.x) >> 5);
  int lane = (int)(threadIdx.x & 31);
  int n4   = N >> 6;                           // 16x64 blocks along N
  int nBlk = (M >> 4) * n4;
  if (wave >= nBlk) return;                    // wave-uniform; EXEC stays all-ones
  int tm = wave / n4, tn = wave % n4;

  int row  = tm * 16 + (lane & 15);
  int half = lane >> 4;                        // 0: lanes 0-15, 1: lanes 16-31
  int klo  = half * 16;                        // A: contiguous 16 K values per lane (matches B)
  int kB   = half * 16;                        // B: contiguous 16 K values per lane
  const __bf16* pA = A + (size_t)row * K;
  const __bf16* pB0 = BT + (size_t)(tn * 64 +  0 + (lane & 15)) * K;
  const __bf16* pB1 = BT + (size_t)(tn * 64 + 16 + (lane & 15)) * K;
  const __bf16* pB2 = BT + (size_t)(tn * 64 + 32 + (lane & 15)) * K;
  const __bf16* pB3 = BT + (size_t)(tn * 64 + 48 + (lane & 15)) * K;

  v8f c0 = {}, c1 = {}, c2 = {}, c3 = {};
  for (int k = 0; k < K; k += 32) {
    v16bf a = *(const v16bf*)(pA + k + klo);       // K = klo..klo+15
    v16bf b0 = *(const v16bf*)(pB0 + k + kB);
    v16bf b1 = *(const v16bf*)(pB1 + k + kB);
    v16bf b2 = *(const v16bf*)(pB2 + k + kB);
    v16bf b3 = *(const v16bf*)(pB3 + k + kB);
    c0 = __builtin_amdgcn_wmma_f32_16x16x32_bf16(false, a, false, b0, (short)0, c0, false, false);
    c1 = __builtin_amdgcn_wmma_f32_16x16x32_bf16(false, a, false, b1, (short)0, c1, false, false);
    c2 = __builtin_amdgcn_wmma_f32_16x16x32_bf16(false, a, false, b2, (short)0, c2, false, false);
    c3 = __builtin_amdgcn_wmma_f32_16x16x32_bf16(false, a, false, b3, (short)0, c3, false, false);
  }

  int m0 = tm * 16 + half * 8;                 // C layout: m = vgpr + 8*(lane>=16)
  int nb = tn * 64 + (lane & 15);
#pragma unroll
  for (int j = 0; j < 4; ++j) {
    int n = nb + j * 16;
    float bv = bias ? bias[n] : 0.0f;
    const v8f& c = (j == 0) ? c0 : (j == 1) ? c1 : (j == 2) ? c2 : c3;
#pragma unroll
    for (int v = 0; v < 8; ++v)
      C[(size_t)(m0 + v) * N + n] = c[v] + bv;
  }
}

// ---------------- conversions ----------------
__global__ void cvt_bf16(const float* __restrict__ src, __bf16* __restrict__ dst, size_t n) {
  size_t i = (size_t)blockIdx.x * blockDim.x + threadIdx.x;
  if (i < n) dst[i] = f2bf(src[i]);
}
// src is [K,N] row-major f32; dst is [N,K] row-major bf16
__global__ void transpose_bf16(const float* __restrict__ src, __bf16* __restrict__ dst,
                               int K, int N) {
  size_t i = (size_t)blockIdx.x * blockDim.x + threadIdx.x;
  if (i >= (size_t)K * N) return;
  int k = (int)(i / N), n = (int)(i % N);
  dst[(size_t)n * K + k] = f2bf(src[(size_t)k * N + n]);
}

// ---------------- per-launch init ----------------
__global__ void init_state(const float* __restrict__ hidden, const float* __restrict__ init_att,
                           const float* __restrict__ coverage, const float* __restrict__ gctx,
                           float* __restrict__ h, __bf16* __restrict__ h_bf,
                           float* __restrict__ cur_ctx, float* __restrict__ cov,
                           __bf16* __restrict__ q_in, __bf16* __restrict__ ro_in) {
  int idx = blockIdx.x * blockDim.x + threadIdx.x;
  if (idx < B * H) {
    float hv = hidden[idx];
    h[idx] = hv; h_bf[idx] = f2bf(hv);
    cur_ctx[idx] = init_att[idx];
    int b = idx >> 9, j = idx & 511;
    __bf16 g = f2bf(gctx[idx]);
    q_in[b * KQ + H + j]  = g;         // global_ctx half of query input (constant)
    ro_in[b * KR + D + H + E + j] = g; // global_ctx slice of readout input
  }
  if (idx < B * S) cov[idx] = coverage[idx];
}

// ---------------- per-step kernels ----------------
__global__ void pack_step(const int* __restrict__ y, const float* __restrict__ embW,
                          const float* __restrict__ cur_ctx,
                          __bf16* __restrict__ x_cat, __bf16* __restrict__ ro_in) {
  int idx = blockIdx.x * blockDim.x + threadIdx.x;     // B*KX
  if (idx >= B * KX) return;
  int b = idx >> 10, d = idx & 1023;
  if (d < D) {
    float e = embW[(size_t)y[b] * D + d];
    __bf16 v = f2bf(e);
    x_cat[b * KX + d] = v;
    ro_in[b * KR + d] = v;
  } else {
    x_cat[b * KX + d] = f2bf(cur_ctx[b * E + (d - D)]);
  }
}

__global__ void gru_combine(const float* __restrict__ gi, const float* __restrict__ gh,
                            float* __restrict__ h, __bf16* __restrict__ h_bf,
                            __bf16* __restrict__ q_in, __bf16* __restrict__ ro_in) {
  int idx = blockIdx.x * blockDim.x + threadIdx.x;     // B*H
  if (idx >= B * H) return;
  int b = idx >> 9, j = idx & 511;
  const float* gib = gi + (size_t)b * G3;
  const float* ghb = gh + (size_t)b * G3;
  float r = sigf(gib[j] + ghb[j]);
  float z = sigf(gib[H + j] + ghb[H + j]);
  float n = tanhf(gib[2 * H + j] + r * ghb[2 * H + j]);
  float h0 = (1.0f - z) * n + z * h[idx];
  h[idx] = h0;
  __bf16 v = f2bf(h0);
  h_bf[idx] = v;
  q_in[b * KQ + j] = v;
  ro_in[b * KR + D + j] = v;
}

// energy[b,s] = sum_a tanh(pre[s,b,a] + q[b,a] + cov[b,s]*Wcov[a]) * Wv[a]; mask pad
__global__ void energy_kernel(const float* __restrict__ pre, const float* __restrict__ q,
                              const float* __restrict__ cov, const float* __restrict__ Wcov,
                              const float* __restrict__ Wv, const float* __restrict__ pad,
                              float* __restrict__ energy) {
  int wave = (int)((blockIdx.x * blockDim.x + threadIdx.x) >> 5);
  int lane = (int)(threadIdx.x & 31);
  if (wave >= B * S) return;
  int b = wave / S, s = wave % S;
  const float* p  = pre + ((size_t)s * B + b) * AA;
  const float* qb = q + (size_t)b * AA;
  float cv = cov[b * S + s];
  float acc = 0.0f;
  int a0 = lane * 16;
#pragma unroll
  for (int i = 0; i < 16; ++i) {
    int a = a0 + i;
    acc += tanhf(p[a] + qb[a] + cv * Wcov[a]) * Wv[a];
  }
#pragma unroll
  for (int o = 16; o > 0; o >>= 1) acc += __shfl_xor(acc, o, 32);
  if (lane == 0) {
    energy[b * S + s] = (pad[b * S + s] > 0.5f) ? -1e18f : acc;
  }
}

__global__ void softmax_cov(const float* __restrict__ energy, float* __restrict__ cov,
                            float* __restrict__ attn_out, float* __restrict__ covloss_out) {
  int b = blockIdx.x, tid = threadIdx.x;               // 512 threads
  __shared__ float red[512];
  float e = (tid < S) ? energy[b * S + tid] : -3.4e38f;
  red[tid] = e; __syncthreads();
  for (int o = 256; o > 0; o >>= 1) { if (tid < o) red[tid] = fmaxf(red[tid], red[tid + o]); __syncthreads(); }
  float mx = red[0]; __syncthreads();
  float ex = (tid < S) ? __expf(e - mx) : 0.0f;
  red[tid] = ex; __syncthreads();
  for (int o = 256; o > 0; o >>= 1) { if (tid < o) red[tid] += red[tid + o]; __syncthreads(); }
  float inv = 1.0f / red[0]; __syncthreads();
  float at = ex * inv, cl = 0.0f;
  if (tid < S) {
    float c = cov[b * S + tid];
    attn_out[b * S + tid] = at;
    cl = fminf(at, c);
    cov[b * S + tid] = c + at;
  }
  red[tid] = cl; __syncthreads();
  for (int o = 256; o > 0; o >>= 1) { if (tid < o) red[tid] += red[tid + o]; __syncthreads(); }
  if (tid == 0) covloss_out[b] = red[0];
}

__global__ void ctx_reduce(const float* __restrict__ attn, const float* __restrict__ context,
                           float* __restrict__ cur_ctx, __bf16* __restrict__ ro_in) {
  int idx = blockIdx.x * blockDim.x + threadIdx.x;     // B*E
  if (idx >= B * E) return;
  int b = idx >> 9, e = idx & 511;
  float acc = 0.0f;
  for (int s = 0; s < S; ++s)
    acc += attn[b * S + s] * context[((size_t)s * B + b) * E + e];
  cur_ctx[idx] = acc;
  ro_in[b * KR + D + H + e] = f2bf(acc);
}

__global__ void pgen_kernel(const float* __restrict__ cur_ctx, const float* __restrict__ h,
                            const float* __restrict__ embW, const int* __restrict__ y,
                            const float* __restrict__ Wpgen, const float* __restrict__ bpgen,
                            float* __restrict__ out) {
  int b = blockIdx.x, tid = threadIdx.x;               // 256 threads
  __shared__ float red[256];
  float acc = 0.0f;
  for (int i = tid; i < E; i += 256) acc += cur_ctx[b * E + i] * Wpgen[i];
  for (int i = tid; i < H; i += 256) acc += h[b * H + i] * Wpgen[E + i];
  int yb = y[b];
  for (int i = tid; i < D; i += 256) acc += embW[(size_t)yb * D + i] * Wpgen[E + H + i];
  red[tid] = acc; __syncthreads();
  for (int o = 128; o > 0; o >>= 1) { if (tid < o) red[tid] += red[tid + o]; __syncthreads(); }
  if (tid == 0) out[b] = sigf(red[0] + bpgen[0]);
}

__global__ void maxout_kernel(const float* __restrict__ ro, float* __restrict__ gout) {
  int idx = blockIdx.x * blockDim.x + threadIdx.x;     // B*HP
  if (idx >= B * HP) return;
  int b = idx >> 8, j = idx & 255;
  gout[idx] = fmaxf(ro[b * H + 2 * j], ro[b * H + 2 * j + 1]);
}

__global__ void finalize_kernel(const float* __restrict__ h, const float* __restrict__ cur_ctx,
                                const float* __restrict__ cov, float* __restrict__ out) {
  int idx = blockIdx.x * blockDim.x + threadIdx.x;
  if (idx < B * H) {
    out[OUT_HID + idx]  = h[idx];
    out[OUT_CCTX + idx] = cur_ctx[idx];
  }
  if (idx < B * S) {
    out[OUT_COVF + idx]  = cov[idx];
    out[OUT_ALAST + idx] = out[OUT_ATTN + (size_t)(T - 1) * B * S + idx];
  }
}

// ---------------- host launcher ----------------
static inline dim3 blocks_for(size_t threads, int bs) { return dim3((unsigned)((threads + bs - 1) / bs)); }

extern "C" void kernel_launch(void* const* d_in, const int* in_sizes, int n_in,
                              void* d_out, int out_size, void* d_ws, size_t ws_size,
                              hipStream_t stream) {
  (void)in_sizes; (void)n_in; (void)out_size; (void)ws_size;
  const int*   input_y  = (const int*)  d_in[0];
  const float* hidden   = (const float*)d_in[1];
  const float* context  = (const float*)d_in[2];
  const float* pad      = (const float*)d_in[3];
  const float* init_att = (const float*)d_in[4];
  const float* coverage = (const float*)d_in[5];
  const float* gctx     = (const float*)d_in[6];
  const float* embW     = (const float*)d_in[7];
  const float* W_ih     = (const float*)d_in[8];
  const float* W_hh     = (const float*)d_in[9];
  const float* b_ih     = (const float*)d_in[10];
  const float* b_hh     = (const float*)d_in[11];
  const float* W_pre    = (const float*)d_in[12];
  const float* b_pre    = (const float*)d_in[13];
  const float* W_q      = (const float*)d_in[14];
  const float* W_v      = (const float*)d_in[15];
  const float* W_cov    = (const float*)d_in[16];
  const float* W_pgen   = (const float*)d_in[17];
  const float* b_pgen   = (const float*)d_in[18];
  const float* W_ro     = (const float*)d_in[19];
  const float* b_ro     = (const float*)d_in[20];
  float* out = (float*)d_out;

  // ---- workspace bump allocator (256B aligned) ----
  char* base = (char*)d_ws;
  size_t off = 0;
  auto alloc = [&](size_t bytes) -> char* {
    char* p = base + off;
    off = (off + bytes + 255) & ~(size_t)255;
    return p;
  };
  float*  pre_f32  = (float*) alloc((size_t)S * B * AA * 4);   // [s,b,a] 52.4 MB (L2-resident)
  __bf16* ctx_bf   = (__bf16*)alloc((size_t)S * B * E * 2);    // 26.2 MB
  __bf16* Wih_bf   = (__bf16*)alloc((size_t)G3 * KX * 2);      // already [N,K]
  __bf16* Whh_bf   = (__bf16*)alloc((size_t)G3 * H * 2);       // already [N,K]
  __bf16* WqT_bf   = (__bf16*)alloc((size_t)AA * KQ * 2);
  __bf16* WpreT_bf = (__bf16*)alloc((size_t)AA * E * 2);
  __bf16* WroT_bf  = (__bf16*)alloc((size_t)H * KR * 2);
  __bf16* x_cat    = (__bf16*)alloc((size_t)B * KX * 2);
  __bf16* h_bf     = (__bf16*)alloc((size_t)B * H * 2);
  __bf16* q_in     = (__bf16*)alloc((size_t)B * KQ * 2);
  __bf16* ro_in    = (__bf16*)alloc((size_t)B * KR * 2);
  float*  h_f32    = (float*) alloc((size_t)B * H * 4);
  float*  cctx_f32 = (float*) alloc((size_t)B * E * 4);
  float*  cov_f32  = (float*) alloc((size_t)B * S * 4);
  float*  gi_f32   = (float*) alloc((size_t)B * G3 * 4);
  float*  gh_f32   = (float*) alloc((size_t)B * G3 * 4);
  float*  q_f32    = (float*) alloc((size_t)B * AA * 4);
  float*  en_f32   = (float*) alloc((size_t)B * S * 4);
  float*  ro_f32   = (float*) alloc((size_t)B * H * 4);

  const int BS = 256;
  auto gemm = [&](const __bf16* A, const __bf16* BT, const float* bias, float* C,
                  int M, int N, int K) {
    size_t blkWaves = (size_t)(M / 16) * (N / 64);   // one wave per 16x64 C block
    gemm_bf16_wmma4<<<blocks_for(blkWaves * 32, BS), BS, 0, stream>>>(A, BT, bias, C, M, N, K);
  };

  // ---- one-time init (per launch; deterministic) ----
  cvt_bf16<<<blocks_for((size_t)G3 * KX, BS), BS, 0, stream>>>(W_ih, Wih_bf, (size_t)G3 * KX);
  cvt_bf16<<<blocks_for((size_t)G3 * H, BS), BS, 0, stream>>>(W_hh, Whh_bf, (size_t)G3 * H);
  cvt_bf16<<<blocks_for((size_t)S * B * E, BS), BS, 0, stream>>>(context, ctx_bf, (size_t)S * B * E);
  transpose_bf16<<<blocks_for((size_t)KQ * AA, BS), BS, 0, stream>>>(W_q, WqT_bf, KQ, AA);
  transpose_bf16<<<blocks_for((size_t)E * AA, BS), BS, 0, stream>>>(W_pre, WpreT_bf, E, AA);
  transpose_bf16<<<blocks_for((size_t)KR * H, BS), BS, 0, stream>>>(W_ro, WroT_bf, KR, H);
  init_state<<<blocks_for((size_t)B * H, BS), BS, 0, stream>>>(
      hidden, init_att, coverage, gctx, h_f32, h_bf, cctx_f32, cov_f32, q_in, ro_in);
  // precompute[s*B+b, a] = ctx @ W_pre + b_pre  (M = S*B = 25600)
  gemm(ctx_bf, WpreT_bf, b_pre, pre_f32, S * B, AA, E);

  // ---- T sequential steps ----
  for (int t = 0; t < T; ++t) {
    const int* y_t     = input_y + (size_t)t * B;
    float* attn_t      = out + OUT_ATTN + (size_t)t * B * S;
    float* covloss_t   = out + OUT_CLOSS + (size_t)t * B;
    float* pg_t        = out + OUT_PG + (size_t)t * B;
    float* gout_t      = out + OUT_GOUT + (size_t)t * B * HP;

    pack_step<<<blocks_for((size_t)B * KX, BS), BS, 0, stream>>>(y_t, embW, cctx_f32, x_cat, ro_in);
    gemm(x_cat, Wih_bf, b_ih, gi_f32, B, G3, KX);
    gemm(h_bf,  Whh_bf, b_hh, gh_f32, B, G3, H);
    gru_combine<<<blocks_for((size_t)B * H, BS), BS, 0, stream>>>(gi_f32, gh_f32, h_f32, h_bf, q_in, ro_in);
    gemm(q_in, WqT_bf, nullptr, q_f32, B, AA, KQ);
    energy_kernel<<<blocks_for((size_t)B * S * 32, BS), BS, 0, stream>>>(
        pre_f32, q_f32, cov_f32, W_cov, W_v, pad, en_f32);
    softmax_cov<<<dim3(B), dim3(512), 0, stream>>>(en_f32, cov_f32, attn_t, covloss_t);
    ctx_reduce<<<blocks_for((size_t)B * E, BS), BS, 0, stream>>>(attn_t, context, cctx_f32, ro_in);
    pgen_kernel<<<dim3(B), dim3(256), 0, stream>>>(cctx_f32, h_f32, embW, y_t, W_pgen, b_pgen, pg_t);
    gemm(ro_in, WroT_bf, b_ro, ro_f32, B, H, KR);
    maxout_kernel<<<blocks_for((size_t)B * HP, BS), BS, 0, stream>>>(ro_f32, gout_t);
  }

  finalize_kernel<<<blocks_for((size_t)B * H, BS), BS, 0, stream>>>(h_f32, cctx_f32, cov_f32, out);
}
